// GraphKHopDecentDiffDistRecompAttentionLayer_76467597738465
// MI455X (gfx1250) — matrix operand order repaired and checked by reading, not compiled, not run on hardware
//
#include <hip/hip_runtime.h>
#include <hip/hip_bf16.h>

#define NN   4096
#define INF_ 512
#define OUTF 256
#define KHOP 3
#define ALPHA 0.2f
#define BETA  0.1f
#define ETA   0.5f
#define NEGINF -9.0e15f

typedef __attribute__((ext_vector_type(16))) __bf16 v16bf;
typedef __attribute__((ext_vector_type(8)))  float  v8f;

struct __align__(16) U8 { unsigned u[8]; };

// round-to-nearest-even f32 -> bf16, packed pair
__device__ __forceinline__ unsigned pack2bf(float lo, float hi) {
    unsigned a = __float_as_uint(lo);
    unsigned b = __float_as_uint(hi);
    a = (a + 0x7FFFu + ((a >> 16) & 1u)) >> 16;
    b = (b + 0x7FFFu + ((b >> 16) & 1u)) >> 16;
    return (a & 0xFFFFu) | (b << 16);
}

__device__ __forceinline__ v8f wmma_bf16(const U8& a, const U8& b, v8f c) {
    v16bf av = __builtin_bit_cast(v16bf, a);
    v16bf bv = __builtin_bit_cast(v16bf, b);
    return __builtin_amdgcn_wmma_f32_16x16x32_bf16(
        /*neg_a=*/false, av, /*neg_b=*/false, bv,
        /*c_mod=*/(short)0, c, /*reuse_a=*/false, /*reuse_b=*/false);
}

// ---------------- pack kernels ----------------
__global__ void k_pack_h(const float* __restrict__ h, unsigned* __restrict__ hB) {
    int i = blockIdx.x * blockDim.x + threadIdx.x;          // pair index
    if (i < NN * INF_ / 2) hB[i] = pack2bf(h[2 * i], h[2 * i + 1]);
}

// Wt[hop][o][k] bf16 (k contiguous) from W[(hop*IN + k)*OUT + o]
__global__ void k_pack_wt(const float* __restrict__ W, unsigned* __restrict__ WtB) {
    int p = blockIdx.x * blockDim.x + threadIdx.x;          // pair index
    if (p >= KHOP * OUTF * INF_ / 2) return;
    int hop = p / (OUTF * INF_ / 2);
    int r   = p % (OUTF * INF_ / 2);
    int o   = r / (INF_ / 2);
    int kk  = (r % (INF_ / 2)) * 2;
    float lo = W[(size_t)(hop * INF_ + kk)     * OUTF + o];
    float hi = W[(size_t)(hop * INF_ + kk + 1) * OUTF + o];
    WtB[p] = pack2bf(lo, hi);
}

// Wh0t[o][j] bf16 (j contiguous) from Wh_all[0][j][o]
__global__ void k_pack_wh0t(const float* __restrict__ Wh_all, unsigned* __restrict__ Wh0t) {
    int p = blockIdx.x * blockDim.x + threadIdx.x;          // pair index
    if (p >= OUTF * NN / 2) return;
    int o = p / (NN / 2);
    int j = (p % (NN / 2)) * 2;
    float lo = Wh_all[(size_t)j       * OUTF + o];
    float hi = Wh_all[(size_t)(j + 1) * OUTF + o];
    Wh0t[p] = pack2bf(lo, hi);
}

// ---------------- GEMM: Wh_all[k] = h @ Wk (bf16 WMMA, f32 acc) ----------------
__global__ __launch_bounds__(256) void k_gemm_wh(const unsigned* __restrict__ hB,
                                                 const unsigned* __restrict__ WtB,
                                                 float* __restrict__ Wh_all) {
    const int MT = NN / 16, NT = OUTF / 16;                 // 256, 16
    int wave = (blockIdx.x * blockDim.x + threadIdx.x) >> 5;
    int lane = threadIdx.x & 31;
    int hop  = wave / (MT * NT);
    int rem  = wave % (MT * NT);
    int mt = rem / NT, nt = rem % NT;

    int rowA  = mt * 16 + (lane & 15);
    int colB  = nt * 16 + (lane & 15);
    int kq    = (lane >> 4) * 8;                            // A fragment k-offset
    int khalf = (lane >> 4) * 16;                           // unused; B loads use kq like A
    const unsigned* aRow = hB  + ((size_t)rowA * INF_) / 2;
    const unsigned* bRow = WtB + ((size_t)(hop * OUTF + colB) * INF_) / 2;

    v8f c = {};
    for (int kb = 0; kb < INF_; kb += 32) {
        U8 a, b;
        *(uint4*)&a.u[0] = *(const uint4*)(aRow + (kb + kq) / 2);
        *(uint4*)&a.u[4] = *(const uint4*)(aRow + (kb + 16 + kq) / 2);
        *(uint4*)&b.u[0] = *(const uint4*)(bRow + (kb + kq) / 2);
        *(uint4*)&b.u[4] = *(const uint4*)(bRow + (kb + 16 + kq) / 2);
        c = wmma_bf16(a, b, c);
    }
    int cn = nt * 16 + (lane & 15);
    float* outp = Wh_all + (size_t)hop * NN * OUTF;
#pragma unroll
    for (int rr = 0; rr < 8; ++rr) {
        int cm = mt * 16 + rr + 8 * (lane >> 4);
        outp[(size_t)cm * OUTF + cn] = c[rr];
    }
}

// ---------------- GEMVs: s1, s2, u, v ----------------
__global__ void k_vecs(const float* __restrict__ Wh_all, const float* __restrict__ a,
                       const float* __restrict__ disc,
                       float* __restrict__ s1, float* __restrict__ s2,
                       float* __restrict__ u,  float* __restrict__ v) {
    int n = blockIdx.x * blockDim.x + threadIdx.x;
    if (n >= NN) return;
    float acc1 = 0.f, acc2 = 0.f, au = 0.f, av = 0.f;
    const float* w0 = Wh_all + (size_t)n * OUTF;
    for (int o = 0; o < OUTF; ++o) {
        float x = w0[o];
        acc1 += x * a[o];
        acc2 += x * a[OUTF + o];
    }
    float wt = 1.0f;
    for (int k = 0; k < KHOP; ++k) {
        const float* wk = Wh_all + ((size_t)k * NN + n) * OUTF;
        float du = 0.f, dv = 0.f;
        for (int o = 0; o < OUTF; ++o) {
            float x = wk[o];
            du += x * disc[o];
            dv += x * disc[OUTF + o];
        }
        au += wt * du; av += wt * dv; wt *= ETA;
    }
    s1[n] = acc1; s2[n] = acc2; u[n] = au; v[n] = av;
}

// ---------------- denom = max(|min u + min v|, max u + max v) ----------------
__global__ __launch_bounds__(1024) void k_denom(const float* __restrict__ u,
                                                const float* __restrict__ v,
                                                float* __restrict__ denom) {
    __shared__ float smnu[1024], smxu[1024], smnv[1024], smxv[1024];
    int t = threadIdx.x;
    float mnu = 3.0e38f, mxu = -3.0e38f, mnv = 3.0e38f, mxv = -3.0e38f;
    for (int i = t; i < NN; i += 1024) {
        float x = u[i]; mnu = fminf(mnu, x); mxu = fmaxf(mxu, x);
        float y = v[i]; mnv = fminf(mnv, y); mxv = fmaxf(mxv, y);
    }
    smnu[t] = mnu; smxu[t] = mxu; smnv[t] = mnv; smxv[t] = mxv;
    __syncthreads();
    for (int s = 512; s > 0; s >>= 1) {
        if (t < s) {
            smnu[t] = fminf(smnu[t], smnu[t + s]);
            smxu[t] = fmaxf(smxu[t], smxu[t + s]);
            smnv[t] = fminf(smnv[t], smnv[t + s]);
            smxv[t] = fmaxf(smxv[t], smxv[t + s]);
        }
        __syncthreads();
    }
    if (t == 0) {
        float mn = smnu[0] + smnv[0];
        float mx = smxu[0] + smxv[0];
        denom[0] = fmaxf(fabsf(mn), mx);
    }
}

// ---------------- per-row softmax stats (reference fills masked with -9e15) ----------
__global__ __launch_bounds__(256) void k_softmax_stats(const float* __restrict__ s1,
                                                       const float* __restrict__ s2,
                                                       const int* __restrict__ mask0,
                                                       float* __restrict__ mrow,
                                                       float* __restrict__ zrow) {
    int i = blockIdx.x;
    int t = threadIdx.x;
    float s1i = s1[i];
    const int* mr = mask0 + (size_t)i * NN;
    float m = -3.0e38f, z = 0.f;
    for (int j = t; j < NN; j += 256) {
        float e = s1i + s2[j];
        e = e > 0.f ? e : ALPHA * e;
        float val = (mr[j] > 0) ? e : NEGINF;
        if (val > m) { z = z * __expf(m - val) + 1.0f; m = val; }
        else          z += __expf(val - m);
    }
    __shared__ float sm[256], sz[256];
    sm[t] = m; sz[t] = z;
    __syncthreads();
    for (int s = 128; s > 0; s >>= 1) {
        if (t < s) {
            float m2 = sm[t + s], z2 = sz[t + s];
            float mm = fmaxf(sm[t], m2);
            sz[t] = sz[t] * __expf(sm[t] - mm) + z2 * __expf(m2 - mm);
            sm[t] = mm;
        }
        __syncthreads();
    }
    if (t == 0) { mrow[i] = sm[0]; zrow[i] = sz[0]; }
}

// ---------------- attention GEMM + epilogue (WMMA) ----------------
// 512 threads = 16 waves per 16-row block. Each wave builds one P fragment
// (bf16, via LDS) per step; all waves consume all 16 fragments against their
// own 16-wide OUT tile.
__global__ __launch_bounds__(512) void k_attn(const unsigned* __restrict__ Wh0t,
                                              const float* __restrict__ Wh0,
                                              const float* __restrict__ s1,
                                              const float* __restrict__ s2,
                                              const float* __restrict__ mrow,
                                              const float* __restrict__ zrow,
                                              const int* __restrict__ mask0,
                                              float* __restrict__ out) {
    __shared__ unsigned lds[16 * 256];                      // 16 fragments, 16KB
    int w    = threadIdx.x >> 5;
    int lane = threadIdx.x & 31;
    int rb   = blockIdx.x;

    int rowA = rb * 16 + (lane & 15);
    int kq   = (lane >> 4) * 8;
    float s1r  = s1[rowA];
    float mr   = mrow[rowA];
    float invZ = 1.0f / zrow[rowA];
    const int* mp = mask0 + (size_t)rowA * NN;

    int ocol  = w * 16 + (lane & 15);
    int khalf = (lane >> 4) * 16;
    const unsigned* bRow = Wh0t + (((size_t)ocol * NN) >> 1);

    v8f c = {};
    for (int t = 0; t < 8; ++t) {
        // produce P fragment for j-tile (t*16 + w)
        int jt = t * 16 + w;
        int j0 = jt * 32 + kq;
        unsigned* dst = &lds[w * 256 + lane * 8];
#pragma unroll
        for (int half = 0; half < 2; ++half) {
            int jb = j0 + half * 16;
#pragma unroll
            for (int d = 0; d < 4; ++d) {
                int j = jb + 2 * d;
                float e0 = s1r + s2[j];
                e0 = e0 > 0.f ? e0 : ALPHA * e0;
                float v0 = (mp[j] > 0) ? e0 : NEGINF;
                float p0 = __expf(v0 - mr) * invZ;
                float e1 = s1r + s2[j + 1];
                e1 = e1 > 0.f ? e1 : ALPHA * e1;
                float v1 = (mp[j + 1] > 0) ? e1 : NEGINF;
                float p1 = __expf(v1 - mr) * invZ;
                dst[half * 4 + d] = pack2bf(p0, p1);
            }
        }
        __syncthreads();
        // consume 16 fragments
#pragma unroll 4
        for (int f = 0; f < 16; ++f) {
            int jtc = t * 16 + f;
            U8 a, b;
            const unsigned* src = &lds[f * 256 + lane * 8];
            *(uint4*)&a.u[0] = *(const uint4*)(src);
            *(uint4*)&a.u[4] = *(const uint4*)(src + 4);
            const unsigned* bp = bRow + (jtc * 32 + kq) / 2;
            *(uint4*)&b.u[0] = *(const uint4*)(bp);
            *(uint4*)&b.u[4] = *(const uint4*)(bp + 8);
            c = wmma_bf16(a, b, c);
        }
        __syncthreads();
    }
    // epilogue: (1-beta)*acc + beta*Wh0, then ELU
    int cn = w * 16 + (lane & 15);
#pragma unroll
    for (int rr = 0; rr < 8; ++rr) {
        int cm = rb * 16 + rr + 8 * (lane >> 4);
        float wh0 = Wh0[(size_t)cm * OUTF + cn];
        float val = (1.0f - BETA) * c[rr] + BETA * wh0;
        out[(size_t)cm * OUTF + cn] = val > 0.f ? val : (__expf(val) - 1.0f);
    }
}

// ---------------- dn + km elementwise (closed-form mask shuffle) ----------------
__global__ void k_dnkm(const float* __restrict__ u, const float* __restrict__ v,
                       const float* __restrict__ denomP, const float* __restrict__ lamP,
                       const int* __restrict__ kmask,
                       float* __restrict__ dn, int* __restrict__ km) {
    size_t idx = (size_t)blockIdx.x * blockDim.x + threadIdx.x;
    if (idx >= (size_t)NN * NN) return;
    int i = (int)(idx >> 12);
    int j = (int)(idx & (NN - 1));
    float denom = denomP[0];
    float lam   = lamP[0];
    float d0 = (u[i] + v[j]) / denom;
    int pos = d0 >  lam ? 1 : 0;
    int neg = d0 < -lam ? 1 : 0;
    dn[idx] = (pos | neg) ? 0.f : d0;

    const size_t P = (size_t)NN * NN;
    int k0 = kmask[idx];
    int k1 = kmask[P + idx];
    int k2 = kmask[2 * P + idx];
    int p0 = k0 & pos, p1 = k1 & pos;
    int n0 = k1 & neg, n1 = k2 & neg;
    km[idx]         = k0 - p0 - n0;
    km[P + idx]     = k1 + p0 - p1 + n0 - n1;
    km[2 * P + idx] = k2 + p1 + n1;
}

// ---------------- host launcher ----------------
extern "C" void kernel_launch(void* const* d_in, const int* in_sizes, int n_in,
                              void* d_out, int out_size, void* d_ws, size_t ws_size,
                              hipStream_t stream) {
    const float* h     = (const float*)d_in[0];
    const float* W     = (const float*)d_in[1];
    const float* a     = (const float*)d_in[2];
    const float* disc  = (const float*)d_in[3];
    const int*   kmask = (const int*)  d_in[4];
    const float* lam   = (const float*)d_in[5];

    // output layout: out [N,OUT] f32 | dn [N,N] f32 | km [K,N,N] i32
    float* out = (float*)d_out;
    float* dn  = out + (size_t)NN * OUTF;
    int*   km  = (int*)(dn + (size_t)NN * NN);

    // workspace layout (≈19.8 MB)
    char* ws = (char*)d_ws;
    size_t off = 0;
    auto take = [&](size_t bytes) { char* p = ws + off; off = (off + bytes + 255) & ~(size_t)255; return p; };
    unsigned* hB     = (unsigned*)take((size_t)NN * INF_ / 2 * 4);          // 4 MB
    unsigned* WtB    = (unsigned*)take((size_t)KHOP * OUTF * INF_ / 2 * 4); // 768 KB
    float*    Wh_all = (float*)   take((size_t)KHOP * NN * OUTF * 4);       // 12 MB
    unsigned* Wh0t   = (unsigned*)take((size_t)OUTF * NN / 2 * 4);          // 2 MB
    float*    s1     = (float*)   take((size_t)NN * 4);
    float*    s2     = (float*)   take((size_t)NN * 4);
    float*    u      = (float*)   take((size_t)NN * 4);
    float*    v      = (float*)   take((size_t)NN * 4);
    float*    mrow   = (float*)   take((size_t)NN * 4);
    float*    zrow   = (float*)   take((size_t)NN * 4);
    float*    denom  = (float*)   take(256);

    k_pack_h  <<<(NN * INF_ / 2 + 255) / 256, 256, 0, stream>>>(h, hB);
    k_pack_wt <<<(KHOP * OUTF * INF_ / 2 + 255) / 256, 256, 0, stream>>>(W, WtB);

    // 3*256*16 = 12288 waves, 8 waves/block
    k_gemm_wh <<<KHOP * (NN / 16) * (OUTF / 16) / 8, 256, 0, stream>>>(hB, WtB, Wh_all);

    k_pack_wh0t <<<(OUTF * NN / 2 + 255) / 256, 256, 0, stream>>>(Wh_all, Wh0t);
    k_vecs      <<<NN / 256, 256, 0, stream>>>(Wh_all, a, disc, s1, s2, u, v);
    k_denom     <<<1, 1024, 0, stream>>>(u, v, denom);
    k_softmax_stats <<<NN, 256, 0, stream>>>(s1, s2, kmask, mrow, zrow);

    k_attn <<<NN / 16, 512, 0, stream>>>(Wh0t, Wh_all, s1, s2, mrow, zrow, kmask, out);

    k_dnkm <<<(unsigned)(((size_t)NN * NN + 255) / 256), 256, 0, stream>>>(
        u, v, denom, lam, kmask, dn, km);
}
